// SpatialCrossAttention_89867895702107
// MI455X (gfx1250) — matrix operand order repaired and checked by reading, not compiled, not run on hardware
//
#include <hip/hip_runtime.h>
#include <stdint.h>

#define EMBEDC 256
#define HEADSC 8
#define LEVELSC 4
#define POINTSC 8
#define CAMSC 6
#define NQC 2500
#define LTOT 19560   // 92*160 + 46*80 + 23*40 + 12*20

typedef __attribute__((ext_vector_type(16))) __bf16          bf16x16;
typedef __attribute__((ext_vector_type(8)))  float           f32x8;
typedef __attribute__((ext_vector_type(4)))  float           f32x4;
typedef __attribute__((ext_vector_type(4)))  unsigned short  u16x4;
typedef __attribute__((ext_vector_type(8)))  unsigned short  u16x8;
typedef __attribute__((ext_vector_type(16))) unsigned short  u16x16;

__device__ __forceinline__ unsigned short f32_to_bf16(float f) {
    unsigned u = __builtin_bit_cast(unsigned, f);
    unsigned r = u + 0x7FFFu + ((u >> 16) & 1u);   // round-to-nearest-even
    return (unsigned short)(r >> 16);
}
__device__ __forceinline__ float bf16_to_f32(unsigned short h) {
    return __builtin_bit_cast(float, (unsigned)h << 16);
}

// ---------------------------------------------------------------------------
// bf16-WMMA GEMM: C[M,N] = A[M,256] * B[256,N] (+bias) (+residual)
// BM=128, BN=256 (full-N blocks -> A read exactly once from HBM), K panels
// of 64 staged through ~55 KB LDS. 256 threads = 8 waves; wave w owns rows
// [w*16, w*16+16) x 16 N-tiles (128 VGPR accumulators). B fragments loaded
// in groups of 4 so WMMAs issue back-to-back behind one ds-wait.
// ---------------------------------------------------------------------------
template<bool OUT_BF16>
__global__ __launch_bounds__(256)
void gemm_bf16_wmma(const float* __restrict__ A, const float* __restrict__ B,
                    const float* __restrict__ bias, const float* __restrict__ residual,
                    void* __restrict__ Cout, int M, int N) {
    constexpr int K = 256, BM = 128, BN = 256, KP = 64;
    constexpr int LDA = KP + 8, LDB = KP + 8, NT = BN / 16;
    __shared__ unsigned short sA [BM][LDA];   // row-major (rows x K-panel)
    __shared__ unsigned short sBt[BN][LDB];   // transposed (cols x K-panel)

    const int tid   = threadIdx.x;
    const int wave  = tid >> 5;
    const int lane  = tid & 31;
    const int l15   = lane & 15;
    const int lhalf = lane >> 4;
    const int rowBase = blockIdx.x * BM;
    const int colBase = blockIdx.y * BN;

    f32x8 acc[NT] = {};

    for (int kp = 0; kp < K; kp += KP) {
        // stage A panel (BM x KP) as bf16, packed ds_store_b64 writes
        for (int i = tid; i < (BM * KP) / 4; i += 256) {
            int r  = i >> 4;            // KP/4 = 16 float4 per row
            int k4 = (i & 15) << 2;
            int gr = rowBase + r; if (gr >= M) gr = M - 1;   // clamp; store is guarded
            f32x4 v = *(const f32x4*)(A + (size_t)gr * K + kp + k4);
            u16x4 p;
            p.x = f32_to_bf16(v.x); p.y = f32_to_bf16(v.y);
            p.z = f32_to_bf16(v.z); p.w = f32_to_bf16(v.w);
            *(u16x4*)&sA[r][k4] = p;
        }
        // stage B panel transposed (BN x KP) as bf16
        for (int i = tid; i < (KP * BN) / 4; i += 256) {
            int k  = i >> 6;            // BN/4 = 64 float4 per k-row
            int n4 = (i & 63) << 2;
            f32x4 v = *(const f32x4*)(B + (size_t)(kp + k) * N + colBase + n4);
            sBt[n4 + 0][k] = f32_to_bf16(v.x);
            sBt[n4 + 1][k] = f32_to_bf16(v.y);
            sBt[n4 + 2][k] = f32_to_bf16(v.z);
            sBt[n4 + 3][k] = f32_to_bf16(v.w);
        }
        // prefetch next A K-panel into cache (global_prefetch_b8)
        if (kp + KP < K) {
            int r  = tid >> 1;
            int kq = (tid & 1) << 5;
            int gr = rowBase + r; if (gr >= M) gr = M - 1;
            __builtin_prefetch(A + (size_t)gr * K + (kp + KP) + kq, 0, 0);
        }
        __syncthreads();

        #pragma unroll
        for (int kb = 0; kb < KP; kb += 32) {
            // A fragment 16x32 bf16 (ISA layout): lanes 0-15 -> K {kb..+7, kb+16..+23},
            // lanes 16-31 -> K {kb+8..+15, kb+24..+31}
            const unsigned short* pa = &sA[wave * 16 + l15][kb + lhalf * 8];
            u16x8 alo = *(const u16x8*)pa;
            u16x8 ahi = *(const u16x8*)(pa + 16);
            bf16x16 afrag = __builtin_bit_cast(bf16x16,
                __builtin_shufflevector(alo, ahi, 0,1,2,3,4,5,6,7,8,9,10,11,12,13,14,15));
            #pragma unroll
            for (int ng = 0; ng < NT; ng += 4) {
                bf16x16 bfrag[4];
                #pragma unroll
                for (int u = 0; u < 4; ++u) {
                    // B fragment 32x16: lanes 0-15 -> K {kb..+7, kb+16..+23}, lanes 16-31 -> K {kb+8..+15, kb+24..+31}
                    const unsigned short* pb = &sBt[(ng + u) * 16 + l15][kb + lhalf * 8];
                    u16x8 blo = *(const u16x8*)pb;
                    u16x8 bhi = *(const u16x8*)(pb + 16);
                    bfrag[u] = __builtin_bit_cast(bf16x16,
                        __builtin_shufflevector(blo, bhi, 0,1,2,3,4,5,6,7,8,9,10,11,12,13,14,15));
                }
                #pragma unroll
                for (int u = 0; u < 4; ++u)
                    acc[ng + u] = __builtin_amdgcn_wmma_f32_16x16x32_bf16(
                        false, afrag, false, bfrag[u], (short)0, acc[ng + u], false, false);
            }
        }
        __syncthreads();
    }

    // C/D layout: lane l15 = column, VGPR j -> row j + 8*lhalf
    #pragma unroll
    for (int nt = 0; nt < NT; ++nt) {
        int gn = colBase + nt * 16 + l15;
        float bv = bias ? bias[gn] : 0.0f;
        #pragma unroll
        for (int j = 0; j < 8; ++j) {
            int gr = rowBase + wave * 16 + j + lhalf * 8;
            if (gr < M) {
                float v = acc[nt][j] + bv;
                if (residual) v += residual[(size_t)gr * N + gn];
                if (OUT_BF16)
                    ((unsigned short*)Cout)[(size_t)gr * N + gn] = f32_to_bf16(v);
                else
                    ((float*)Cout)[(size_t)gr * N + gn] = v;
            }
        }
    }
}

// ---------------------------------------------------------------------------
// Deformable sampling: one block per (query, camera); 256 threads = 8 waves.
// Early-exits whole block when bev_mask says this (q,cam) is invalid (the
// reference zero-multiplies those; ~24% of blocks skip all gather work).
// Phase A: wave h does softmax over its 32 (level,point) slots via shuffles,
// then bilinear weights/row indices into LDS. Phase B: lane d of wave h
// accumulates channel h*32+d over 32 slots x 4 corners from the bf16 value
// tensor (60 MB -> resident in the 192 MB L2).
// ---------------------------------------------------------------------------
__global__ __launch_bounds__(256)
void deform_sample(const float* __restrict__ off_buf,        // (NQ, 512)
                   const float* __restrict__ logit_buf,      // (NQ, 256)
                   const float* __restrict__ ref,            // (CAMS,1,NQ,4,2)
                   const unsigned char* __restrict__ bev,    // (CAMS,1,NQ,4) bool
                   const unsigned short* __restrict__ val,   // (CAMS*LTOT, 256) bf16
                   float* __restrict__ out_cam) {             // (CAMS, NQ, 256)
    const int q   = blockIdx.x;
    const int cam = blockIdx.y;

    // uniform early exit before any barrier: invalid camera for this query
    {
        const size_t mb = ((size_t)cam * NQC + q) * 4;
        if (!(bev[mb] | bev[mb + 1] | bev[mb + 2] | bev[mb + 3])) return;
    }

    const int t   = threadIdx.x;
    const int h   = t >> 5;
    const int j   = t & 31;          // = lane
    const int lvl = j >> 3, pt = j & 7;

    __shared__ float sm_cw[256 * 4];
    __shared__ int   sm_row[256 * 4];

    const int Hs[4] = {92, 46, 23, 12};
    const int Ws[4] = {160, 80, 40, 20};
    const int St[4] = {0, 14720, 18400, 19320};

    // softmax over the wave's 32 slots
    float lg = logit_buf[(size_t)q * 256 + h * 32 + j];
    float mx = lg;
    #pragma unroll
    for (int s = 16; s > 0; s >>= 1) mx = fmaxf(mx, __shfl_xor(mx, s, 32));
    float e = __expf(lg - mx);
    float sum = e;
    #pragma unroll
    for (int s = 16; s > 0; s >>= 1) sum += __shfl_xor(sum, s, 32);
    float aw = e / sum;

    const int Wl = Ws[lvl], Hl = Hs[lvl];
    const size_t oidx = (size_t)q * 512 + (((h * LEVELSC + lvl) * POINTSC + pt) * 2);
    const float ox = off_buf[oidx + 0];
    const float oy = off_buf[oidx + 1];
    const int z = pt & 3;                       // p = pp*Z + z
    const size_t ridx = (((size_t)cam * NQC + q) * 4 + z) * 2;
    // x = (ref + off/W)*W - 0.5 = ref*W + off - 0.5
    const float x = ref[ridx + 0] * (float)Wl + ox - 0.5f;
    const float y = ref[ridx + 1] * (float)Hl + oy - 0.5f;
    const float x0 = floorf(x), y0 = floorf(y);
    const float fx = x - x0, fy = y - y0;
    const int xi0 = (int)x0, yi0 = (int)y0;
    #pragma unroll
    for (int c = 0; c < 4; ++c) {
        const int dx = c & 1, dy = c >> 1;
        const int xi = xi0 + dx, yi = yi0 + dy;
        const bool valid = (xi >= 0) && (xi < Wl) && (yi >= 0) && (yi < Hl);
        const float wgt = (dx ? fx : 1.0f - fx) * (dy ? fy : 1.0f - fy);
        const int xc = xi < 0 ? 0 : (xi > Wl - 1 ? Wl - 1 : xi);
        const int yc = yi < 0 ? 0 : (yi > Hl - 1 ? Hl - 1 : yi);
        sm_cw [t * 4 + c] = valid ? wgt * aw : 0.0f;
        sm_row[t * 4 + c] = cam * LTOT + St[lvl] + yc * Wl + xc;
    }
    __syncthreads();

    const int d = t & 31;
    float acc = 0.0f;
    #pragma unroll 4
    for (int jj = 0; jj < 32; ++jj) {
        const int base = (h * 32 + jj) * 4;
        #pragma unroll
        for (int c = 0; c < 4; ++c) {
            const float w = sm_cw[base + c];     // wave-uniform -> cheap branch
            if (w != 0.0f) {
                const int row = sm_row[base + c];
                acc = fmaf(w, bf16_to_f32(val[(size_t)row * 256 + h * 32 + d]), acc);
            }
        }
    }
    out_cam[(((size_t)cam * NQC + q) * 256) + h * 32 + d] = acc;
}

// ---------------------------------------------------------------------------
// Per-query masked camera average. Only reads out_cam for valid cams (invalid
// blocks were skipped and left their out_cam slice unwritten).
// ---------------------------------------------------------------------------
__global__ __launch_bounds__(256)
void combine_cams(const float* __restrict__ out_cam,
                  const unsigned char* __restrict__ bev,        // (CAMS,1,NQ,4) bool
                  float* __restrict__ slots) {                  // (NQ, 256)
    const int q = blockIdx.x, t = threadIdx.x;
    float cnt = 0.0f, s = 0.0f;
    #pragma unroll
    for (int cam = 0; cam < CAMSC; ++cam) {
        const size_t mb = ((size_t)cam * NQC + q) * 4;
        if (bev[mb] | bev[mb + 1] | bev[mb + 2] | bev[mb + 3]) {
            cnt += 1.0f;
            s += out_cam[((size_t)cam * NQC + q) * 256 + t];
        }
    }
    slots[(size_t)q * 256 + t] = s / fmaxf(cnt, 1.0f);
}

// ---------------------------------------------------------------------------
extern "C" void kernel_launch(void* const* d_in, const int* in_sizes, int n_in,
                              void* d_out, int out_size, void* d_ws, size_t ws_size,
                              hipStream_t stream) {
    const float* query  = (const float*)d_in[0];              // (1,2500,256)
    const float* keyf   = (const float*)d_in[1];              // (6,19560,1,256)
    const float* refpts = (const float*)d_in[2];              // (6,1,2500,4,2)
    const unsigned char* bev = (const unsigned char*)d_in[3]; // (6,1,2500,4) bool
    // d_in[4] spatial_shapes: baked as constants
    const float* W_value = (const float*)d_in[5];
    const float* b_value = (const float*)d_in[6];
    const float* W_off   = (const float*)d_in[7];
    const float* b_off   = (const float*)d_in[8];
    const float* W_attn  = (const float*)d_in[9];
    const float* b_attn  = (const float*)d_in[10];
    const float* W_out   = (const float*)d_in[11];
    const float* b_out   = (const float*)d_in[12];

    // workspace layout (~86 MB)
    char* ws = (char*)d_ws;
    size_t o = 0;
    unsigned short* val = (unsigned short*)(ws + o); o += (size_t)CAMSC * LTOT * 256 * 2;
    float* off_buf   = (float*)(ws + o); o += (size_t)NQC * 512 * 4;
    float* logit_buf = (float*)(ws + o); o += (size_t)NQC * 256 * 4;
    float* out_cam   = (float*)(ws + o); o += (size_t)CAMSC * NQC * 256 * 4;
    float* slots     = (float*)(ws + o); o += (size_t)NQC * 256 * 4;
    (void)ws_size; (void)in_sizes; (void)n_in; (void)out_size;

    // 1) value projection: (117360,256)@(256,256) -> bf16 value tensor.
    //    BN=256 -> key_feats (120 MB) read exactly once from HBM.
    {
        const int M = CAMSC * LTOT;
        dim3 g((M + 127) / 128, 1);
        gemm_bf16_wmma<true><<<g, 256, 0, stream>>>(keyf, W_value, b_value, nullptr,
                                                    (void*)val, M, 256);
    }
    // 2) sampling offsets: q @ W_off (+b) -> (2500,512)  [camera-independent]
    {
        dim3 g((NQC + 127) / 128, 512 / 256);
        gemm_bf16_wmma<false><<<g, 256, 0, stream>>>(query, W_off, b_off, nullptr,
                                                     (void*)off_buf, NQC, 512);
    }
    // 3) attention logits: q @ W_attn (+b) -> (2500,256)
    {
        dim3 g((NQC + 127) / 128, 1);
        gemm_bf16_wmma<false><<<g, 256, 0, stream>>>(query, W_attn, b_attn, nullptr,
                                                     (void*)logit_buf, NQC, 256);
    }
    // 4) deformable bilinear sampling per (query, camera), invalid blocks skip
    {
        dim3 g(NQC, CAMSC);
        deform_sample<<<g, 256, 0, stream>>>(off_buf, logit_buf, refpts, bev, val, out_cam);
    }
    // 5) masked camera average
    combine_cams<<<NQC, 256, 0, stream>>>(out_cam, bev, slots);

    // 6) output projection + bias + residual -> d_out
    {
        dim3 g((NQC + 127) / 128, 1);
        gemm_bf16_wmma<false><<<g, 256, 0, stream>>>(slots, W_out, b_out, query,
                                                     d_out, NQC, 256);
    }
}
